// Attention_15075335209532
// MI455X (gfx1250) — matrix operand order repaired and checked by reading, not compiled, not run on hardware
//
#include <hip/hip_runtime.h>
#include <hip/hip_bf16.h>

typedef __attribute__((ext_vector_type(16))) __bf16 v16bf;
typedef __attribute__((ext_vector_type(8)))  float  v8f;

#define BB 4
#define TT 8192
#define DD 512
#define CT 128                 // scan chunk length
#define NC (TT / CT)           // 64 chunks per batch row

// Fast tanh: prefer gfx1250 hardware v_tanh_f32; fallback is branchless
// tanh(x) = 1 - 2/(e^{2x}+1) (correct at +/-inf: exp->inf => 1, exp->0 => -1).
__device__ __forceinline__ float fast_tanh(float x) {
#if __has_builtin(__builtin_amdgcn_tanhf)
    return __builtin_amdgcn_tanhf(x);
#elif __has_builtin(__builtin_amdgcn_tanh_f32)
    return __builtin_amdgcn_tanh_f32(x);
#else
    const float ex = __expf(2.0f * x);
    return fmaf(-2.0f, __builtin_amdgcn_rcpf(ex + 1.0f), 1.0f);
#endif
}

// ---------------------------------------------------------------------------
// Kernel 0: pack W1 (f32 [N=512][K=512], row-major, W1[e][d]) into bf16 in the
// per-lane B-fragment order this file uses for v_wmma_f32_16x16x32_bf16 (contiguous k, same as A):
//   flat = ((ntile*16 + kk)*32 + lane)*16 + elem
//   lane: ncol = lane&15 (N%16), half = lane>>4
//   elem pair j (elem=2j,2j+1): K = kk*32 + 16*half + 2*j (+1)
// so each lane's 16 bf16 elements are 32 contiguous bytes; for a fixed lane,
// kk stride = 1024 B and ntile stride = 16384 B (both fit GLOBAL IOFFSET).
// ---------------------------------------------------------------------------
__global__ void packW_kernel(const float* __restrict__ W1,
                             unsigned short* __restrict__ Wb) {
    int p = blockIdx.x * 256 + threadIdx.x;        // 0 .. 512*512-1
    int e    =  p        & 15;
    int lane = (p >> 4)  & 31;
    int kk   = (p >> 9)  & 15;
    int nt   =  p >> 13;
    int half = lane >> 4, ncol = lane & 15;
    int j = e >> 1, r = e & 1;
    int K = kk * 32 + 16 * half + 2 * j + r;
    int N = nt * 16 + ncol;
    union { __bf16 h; unsigned short u; } cv;
    cv.h = (__bf16)W1[N * DD + K];
    Wb[p] = cv.u;
}

// ---------------------------------------------------------------------------
// Kernel 1: scores. One workgroup (256 thr = 8 waves) per 16-row tile of x.
// x tile staged in LDS as bf16. Loop nest: kk outer, 4 N-tiles inner with 4
// live accumulators, so each A fragment is loaded from LDS once and feeds 4
// back-to-back WMMAs. B fragments stream from one base address + immediate
// offsets (q*16384 + kk*1024). Live VGPRs ~ 4x8 acc + 8 A + B in flight: no
// spills. Then s[m] = sum_N tanh(h)*w2 + b2 via shfl_xor + LDS reduction.
// ---------------------------------------------------------------------------
__global__ void scores_kernel(const float* __restrict__ x,
                              const unsigned short* __restrict__ Wb,
                              const float* __restrict__ b1,
                              const float* __restrict__ w2,
                              const float* __restrict__ b2p,
                              float* __restrict__ s) {
    __shared__ __bf16 xs[16 * DD];        // 16 KB
    __shared__ float  sred[8 * 16];       // per-wave partial s[m]

    const int tile = blockIdx.x;                    // 0 .. B*T/16-1
    const int b    = tile / (TT / 16);
    const int t0   = (tile % (TT / 16)) * 16;
    const int tid  = threadIdx.x;

    // stage x tile (f32 -> bf16) into LDS
    const float* xin = x + (size_t)(b * TT + t0) * DD;
    for (int i = tid; i < 16 * DD; i += 256) xs[i] = (__bf16)xin[i];
    __syncthreads();

    const int wave = tid >> 5;
    const int lane = tid & 31;
    const int half = lane >> 4;
    const int ncol = lane & 15;
    const int m    = ncol;                 // A-fragment row for this lane

    // A fragment row base in LDS: remaining index is a pure constant per (kk,j)
    const __bf16* arow = xs + m * DD + 16 * half;

    // B base for this wave's 4 N-tiles (nt = wave*4 + q):
    // byte offset = ((nt*16+kk)*32+lane)*32 = base + q*16384 + kk*1024
    const char* bbase = (const char*)Wb + ((size_t)(wave * 4) * 512 + lane) * 32;

    // 4 accumulators, one per N-tile, initialized with the bias row
    v8f c[4];
#pragma unroll
    for (int q = 0; q < 4; ++q) {
        const float bias = b1[(wave * 4 + q) * 16 + ncol];
#pragma unroll
        for (int r = 0; r < 8; ++r) c[q][r] = bias;
    }

#pragma unroll
    for (int kk = 0; kk < 16; ++kk) {
        // A fragment in the same contiguous k order as the packed B:
        //   elem pair j: K = kk*32 + 2*j + 16*half
        v16bf afrag;
#pragma unroll
        for (int j = 0; j < 8; ++j) {
            const int K = kk * 32 + 2 * j;   // + 16*half in arow
            afrag[2 * j]     = arow[K];
            afrag[2 * j + 1] = arow[K + 1];
        }
#pragma unroll
        for (int q = 0; q < 4; ++q) {
            const v16bf bfrag =
                *reinterpret_cast<const v16bf*>(bbase + q * 16384 + kk * 1024);
            c[q] = __builtin_amdgcn_wmma_f32_16x16x32_bf16(
                       false, afrag, false, bfrag, (short)0, c[q], false, false);
        }
    }

    float partial[8];
#pragma unroll
    for (int r = 0; r < 8; ++r) partial[r] = 0.f;
#pragma unroll
    for (int q = 0; q < 4; ++q) {
        const float wn = w2[(wave * 4 + q) * 16 + ncol];
#pragma unroll
        for (int r = 0; r < 8; ++r) partial[r] += fast_tanh(c[q][r]) * wn;
    }

    // reduce over the 16 lanes of each half (xor 1,2,4,8 stays within a half)
#pragma unroll
    for (int r = 0; r < 8; ++r) {
        partial[r] += __shfl_xor(partial[r], 1, 32);
        partial[r] += __shfl_xor(partial[r], 2, 32);
        partial[r] += __shfl_xor(partial[r], 4, 32);
        partial[r] += __shfl_xor(partial[r], 8, 32);
    }
    // lane 0 holds m=0..7 (M=r), lane 16 holds m=8..15 (M=8+r)
    if (ncol == 0) {
#pragma unroll
        for (int r = 0; r < 8; ++r) sred[wave * 16 + half * 8 + r] = partial[r];
    }
    __syncthreads();
    if (tid < 16) {
        float tot = 0.f;
#pragma unroll
        for (int w = 0; w < 8; ++w) tot += sred[w * 16 + tid];
        s[b * TT + t0 + tid] = tot + b2p[0];
    }
}

// ---------------------------------------------------------------------------
// Kernel 2: per-batch global max of s
// ---------------------------------------------------------------------------
__global__ void max_kernel(const float* __restrict__ s, float* __restrict__ mvec) {
    __shared__ float red[256];
    const int b = blockIdx.x, tid = threadIdx.x;
    float mx = -__builtin_inff();
    for (int t = tid; t < TT; t += 256) mx = fmaxf(mx, s[b * TT + t]);
    red[tid] = mx;
    __syncthreads();
    for (int o = 128; o > 0; o >>= 1) {
        if (tid < o) red[tid] = fmaxf(red[tid], red[tid + o]);
        __syncthreads();
    }
    if (tid == 0) mvec[b] = red[0];
}

// ---------------------------------------------------------------------------
// Kernel 3: per-chunk partial sums of e and e*x. Block = (b,chunk), 512 thr
// (one per d). e staged once in LDS, shared by all 512 d-lanes.
// ---------------------------------------------------------------------------
__global__ void chunksum_kernel(const float* __restrict__ x,
                                const float* __restrict__ s,
                                const float* __restrict__ mvec,
                                float* __restrict__ exsum,
                                float* __restrict__ esum) {
    __shared__ float elds[CT];
    const int blk = blockIdx.x;            // 0 .. B*NC-1
    const int b   = blk / NC;
    const int t0  = (blk % NC) * CT;
    const int tid = threadIdx.x;
    const float mb = mvec[b];
    if (tid < CT) elds[tid] = __expf(s[b * TT + t0 + tid] - mb);
    __syncthreads();
    float acc = 0.f;
    const float* xp = x + (size_t)(b * TT + t0) * DD + tid;
#pragma unroll 4
    for (int t = 0; t < CT; ++t) acc = fmaf(elds[t], xp[(size_t)t * DD], acc);
    exsum[(size_t)blk * DD + tid] = acc;
    if (tid == 0) {
        float es = 0.f;
        for (int t = 0; t < CT; ++t) es += elds[t];
        esum[blk] = es;
    }
}

// ---------------------------------------------------------------------------
// Kernel 4: exclusive scan over chunks (tiny: 4 blocks x 512 thr, 64 steps)
// ---------------------------------------------------------------------------
__global__ void scan_kernel(const float* __restrict__ exsum,
                            const float* __restrict__ esum,
                            float* __restrict__ xoff,
                            float* __restrict__ eoff) {
    const int b = blockIdx.x, tid = threadIdx.x;
    float run = 0.f;
    for (int c = 0; c < NC; ++c) {
        const size_t i = ((size_t)(b * NC + c)) * DD + tid;
        xoff[i] = run;
        run += exsum[i];
    }
    if (tid == 0) {
        float r = 0.f;
        for (int c = 0; c < NC; ++c) { eoff[b * NC + c] = r; r += esum[b * NC + c]; }
    }
}

// ---------------------------------------------------------------------------
// Kernel 5: final running sums + normalize. Block = (b,chunk), thread = d.
// Normalization uses v_rcp_f32 (1-ulp) instead of the full IEEE div sequence.
// ---------------------------------------------------------------------------
__global__ void final_kernel(const float* __restrict__ x,
                             const float* __restrict__ s,
                             const float* __restrict__ mvec,
                             const float* __restrict__ xoff,
                             const float* __restrict__ eoff,
                             float* __restrict__ out) {
    __shared__ float elds[CT];
    const int blk = blockIdx.x;
    const int b   = blk / NC;
    const int t0  = (blk % NC) * CT;
    const int tid = threadIdx.x;
    const float mb = mvec[b];
    if (tid < CT) elds[tid] = __expf(s[b * TT + t0 + tid] - mb);
    __syncthreads();
    float run = xoff[(size_t)blk * DD + tid];
    float den = eoff[blk];
    const size_t base = (size_t)(b * TT + t0) * DD + tid;
    for (int t = 0; t < CT; ++t) {
        const float e = elds[t];
        run = fmaf(e, x[base + (size_t)t * DD], run);
        den += e;
        out[base + (size_t)t * DD] = run * __builtin_amdgcn_rcpf(den);
    }
}

// ---------------------------------------------------------------------------
extern "C" void kernel_launch(void* const* d_in, const int* in_sizes, int n_in,
                              void* d_out, int out_size, void* d_ws, size_t ws_size,
                              hipStream_t stream) {
    (void)in_sizes; (void)n_in; (void)out_size; (void)ws_size;
    const float* x   = (const float*)d_in[0];   // [B,T,D]
    const float* W1  = (const float*)d_in[1];   // [D,D]
    const float* b1  = (const float*)d_in[2];   // [D]
    const float* w2  = (const float*)d_in[3];   // [D]
    const float* b2  = (const float*)d_in[4];   // [1]
    float* out = (float*)d_out;                 // [B,T,D]

    // workspace layout
    char* ws = (char*)d_ws;
    const size_t WB_BYTES    = (size_t)DD * DD * 2;           // 512 KB packed bf16 W
    const size_t S_BYTES     = (size_t)BB * TT * 4;           // 128 KB scores
    const size_t M_BYTES     = 256;                           // per-batch max (padded)
    const size_t ESUM_BYTES  = (size_t)BB * NC * 4;           // chunk e sums
    const size_t EOFF_BYTES  = (size_t)BB * NC * 4;           // chunk e offsets
    const size_t EXSUM_BYTES = (size_t)BB * NC * DD * 4;      // 512 KB chunk e*x sums
    unsigned short* Wb = (unsigned short*)ws;                 ws += WB_BYTES;
    float* sbuf  = (float*)ws;                                ws += S_BYTES;
    float* mvec  = (float*)ws;                                ws += M_BYTES;
    float* esum  = (float*)ws;                                ws += ESUM_BYTES;
    float* eoff  = (float*)ws;                                ws += EOFF_BYTES;
    float* exsum = (float*)ws;                                ws += EXSUM_BYTES;
    float* xoff  = (float*)ws;

    // 0) pack W1 -> bf16 fragment order
    packW_kernel<<<(DD * DD) / 256, 256, 0, stream>>>(W1, Wb);
    // 1) WMMA score GEMM + tanh + dot(w2)
    scores_kernel<<<(BB * TT) / 16, 256, 0, stream>>>(x, Wb, b1, w2, b2, sbuf);
    // 2) per-batch max
    max_kernel<<<BB, 256, 0, stream>>>(sbuf, mvec);
    // 3) per-chunk partial sums of e, e*x
    chunksum_kernel<<<BB * NC, DD, 0, stream>>>(x, sbuf, mvec, exsum, esum);
    // 4) exclusive scan over chunks
    scan_kernel<<<BB, DD, 0, stream>>>(exsum, esum, xoff, eoff);
    // 5) in-chunk running sums + normalize
    final_kernel<<<BB * NC, DD, 0, stream>>>(x, sbuf, mvec, xoff, eoff, out);
}
